// Self_Attention_16449724745073
// MI455X (gfx1250) — matrix operand order repaired and checked by reading, not compiled, not run on hardware
//
#include <hip/hip_runtime.h>
#include <hip/hip_bf16.h>

#define HEADS  16
#define HSIZE  64
#define DMODEL 1024
#define BATCH  8
#define SEQ    1024

typedef __attribute__((ext_vector_type(16))) __bf16 bf16x16;
typedef __attribute__((ext_vector_type(8)))  __bf16 bf16x8;
typedef __attribute__((ext_vector_type(8)))  float  f32x8;

// ---------- helpers ----------

__device__ __forceinline__ unsigned short f2bf(float f) {
    unsigned u = __float_as_uint(f);
    u += 0x7FFFu + ((u >> 16) & 1u);   // round-to-nearest-even
    return (unsigned short)(u >> 16);
}

// Build a 16-element bf16 fragment from two 16-byte contiguous chunks.
// Works for both global and LDS pointers (compiler infers addrspace).
__device__ __forceinline__ bf16x16 load_frag16(const unsigned short* p0,
                                               const unsigned short* p1) {
    bf16x8 a = *reinterpret_cast<const bf16x8*>(p0);
    bf16x8 b = *reinterpret_cast<const bf16x8*>(p1);
    bf16x16 r;
#pragma unroll
    for (int i = 0; i < 8; ++i) { r[i] = a[i]; r[i + 8] = b[i]; }
    return r;
}

__device__ __forceinline__ f32x8 wmma_bf16(bf16x16 a, bf16x16 b, f32x8 c) {
    return __builtin_amdgcn_wmma_f32_16x16x32_bf16(
        false, a, false, b, (short)0, c, false, false);
}

// ---------- stage 0: precision conversion / layout transforms ----------

__global__ void cvt_inputs(const float* __restrict__ x,
                           unsigned short* __restrict__ xb, int n) {
    int i = blockIdx.x * blockDim.x + threadIdx.x;
    int stride = gridDim.x * blockDim.x;
    for (; i < n; i += stride) xb[i] = f2bf(x[i]);
}

// w: [H,3,D,S] fp32  ->  wt: [H,3,S,D] bf16 (d contiguous for B-fragments)
__global__ void cvt_weights(const float* __restrict__ w,
                            unsigned short* __restrict__ wt) {
    const int total = HEADS * 3 * HSIZE * DMODEL;
    int i = blockIdx.x * blockDim.x + threadIdx.x;
    int stride = gridDim.x * blockDim.x;
    for (; i < total; i += stride) {
        int d  = i % DMODEL;
        int s  = (i / DMODEL) % HSIZE;
        int hj = i / (DMODEL * HSIZE);
        wt[i] = f2bf(w[((size_t)hj * DMODEL + d) * HSIZE + s]);
    }
}

// ---------- stage 1: fused QKV projection (M=8192, K=1024, N=3072) ----------
// One wave computes a 32x64 register-blocked tile: per k-step of 32,
// 2 A-fragments + 4 B-fragments feed 8 WMMAs (0.75 KB loaded per WMMA).
// A 64-wide col tile stays inside one (h, j) weight slice -> uniform stores.
// Q,K stored [b,h,n,s] bf16;  V stored transposed [b,h,s,n] bf16.

__global__ __launch_bounds__(256)
void qkv_gemm(const unsigned short* __restrict__ xb,   // [B*N, D] bf16
              const unsigned short* __restrict__ wt,   // [H,3,S,D] bf16
              unsigned short* __restrict__ Qo,
              unsigned short* __restrict__ Ko,
              unsigned short* __restrict__ Vt) {
    const int lane = threadIdx.x & 31;
    const int wave = threadIdx.x >> 5;
    const int tile = blockIdx.x * 8 + wave;        // 256 * 48 = 12288 tiles
    const int ct   = tile % 48;                    // 3072/64 col groups
    const int rt   = tile / 48;                    // 8192/32 row groups
    const int c0   = ct * 64;
    const int row0 = rt * 32;

    const int half = lane >> 4;
    const int l16  = lane & 15;

    const unsigned short* arow0 = xb + (size_t)(row0 + l16)      * DMODEL;
    const unsigned short* arow1 = xb + (size_t)(row0 + 16 + l16) * DMODEL;

    f32x8 acc[2][4] = {};
    for (int k0 = 0; k0 < DMODEL; k0 += 32) {
        // A 16x32: lane half selects K range {0..15, 16..31}, contiguous
        bf16x16 a0 = load_frag16(arow0 + k0 + half * 16,
                                 arow0 + k0 + half * 16 + 8);
        bf16x16 a1 = load_frag16(arow1 + k0 + half * 16,
                                 arow1 + k0 + half * 16 + 8);
#pragma unroll
        for (int bc = 0; bc < 4; ++bc) {
            // B 32x16: lane half selects K range {0..15, 16..31}, contiguous
            const unsigned short* br =
                wt + (size_t)(c0 + bc * 16 + l16) * DMODEL + k0 + half * 16;
            bf16x16 bfr = load_frag16(br, br + 8);
            acc[0][bc] = wmma_bf16(a0, bfr, acc[0][bc]);
            acc[1][bc] = wmma_bf16(a1, bfr, acc[1][bc]);
        }
    }

    // 64 | col tile -> single (h, j) slice, s0 = 0
    const int h    = ct / 3;
    const int j    = ct % 3;
    const int bidx = row0 >> 10;                   // 1024 rows per batch
    const int n0   = row0 & 1023;
    const size_t bh = (size_t)bidx * HEADS + h;

    if (j < 2) {
        unsigned short* dst = (j == 0 ? Qo : Ko) + bh * SEQ * HSIZE;
#pragma unroll
        for (int ar = 0; ar < 2; ++ar) {
#pragma unroll
            for (int bc = 0; bc < 4; ++bc) {
                const int scol = bc * 16 + l16;
#pragma unroll
                for (int r = 0; r < 8; ++r) {
                    const int n = n0 + ar * 16 + r + half * 8;
                    dst[(size_t)n * HSIZE + scol] = f2bf(acc[ar][bc][r]);
                }
            }
        }
    } else {
        unsigned short* dst = Vt + bh * HSIZE * SEQ;   // [s][n]
#pragma unroll
        for (int ar = 0; ar < 2; ++ar) {
#pragma unroll
            for (int bc = 0; bc < 4; ++bc) {
                const int scol = bc * 16 + l16;
#pragma unroll
                for (int r = 0; r < 8; ++r) {
                    const int n = n0 + ar * 16 + r + half * 8;
                    dst[(size_t)scol * SEQ + n] = f2bf(acc[ar][bc][r]);
                }
            }
        }
    }
}

// ---------- stage 2: flash attention ----------
// One wave per (b, h, 16-query tile). Streams key chunks of 32,
// online softmax, P round-trips through per-wave LDS to become an A-fragment.
// The 8 waves of a block share (b,h) with consecutive query tiles, so their
// K/V chunk loads coincide in the WGP cache.

__global__ __launch_bounds__(256)
void attn(const unsigned short* __restrict__ Q,     // [b,h,n,s] bf16
          const unsigned short* __restrict__ K,     // [b,h,n,s] bf16
          const unsigned short* __restrict__ Vt,    // [b,h,s,n] bf16
          const float* __restrict__ mask,           // [B,N]
          float* __restrict__ out) {                // [B,N, s*H + h]
    __shared__ unsigned short Plds[8][16 * 32];     // 1KB per wave

    const int lane = threadIdx.x & 31;
    const int wave = threadIdx.x >> 5;
    const int id   = blockIdx.x * 8 + wave;         // 8192 tiles
    const int qt   = id & 63;
    const int h    = (id >> 6) & 15;
    const int bidx = id >> 10;
    const int half = lane >> 4;
    const int l16  = lane & 15;
    const int n0   = qt * 16;

    const size_t bh = (size_t)bidx * HEADS + h;
    const unsigned short* Qb = Q  + bh * SEQ * HSIZE;
    const unsigned short* Kb = K  + bh * SEQ * HSIZE;
    const unsigned short* Vb = Vt + bh * HSIZE * SEQ;
    const float* maskb = mask + (size_t)bidx * SEQ;

    // Q A-fragments for s=0..31 and s=32..63, kept in registers all loop
    const unsigned short* qrow = Qb + (size_t)(n0 + l16) * HSIZE;
    const bf16x16 qa0 = load_frag16(qrow +      half * 16, qrow +  8 + half * 16);
    const bf16x16 qa1 = load_frag16(qrow + 32 + half * 16, qrow + 40 + half * 16);

    float mrow[8];
#pragma unroll
    for (int r = 0; r < 8; ++r) mrow[r] = maskb[n0 + r + half * 8];

    f32x8 o0 = {}, o1 = {}, o2 = {}, o3 = {};
    float mmax[8], lsum[8];
#pragma unroll
    for (int r = 0; r < 8; ++r) { mmax[r] = -3.0e38f; lsum[r] = 0.0f; }

    unsigned short* pl = &Plds[wave][0];

    for (int it = 0; it < SEQ / 32; ++it) {
        const int m0 = it * 32;

        // ---- scores for two 16-key tiles (k over s = 64 -> 2 WMMAs each)
        const unsigned short* kr0 = Kb + (size_t)(m0 + l16) * HSIZE + half * 16;
        const unsigned short* kr1 = Kb + (size_t)(m0 + 16 + l16) * HSIZE + half * 16;
        // prefetch next key chunk (global_prefetch_b8)
        __builtin_prefetch(kr0 + 32 * HSIZE, 0, 1);
        f32x8 s0 = {}, s1 = {};
        {
            bf16x16 kf = load_frag16(kr0, kr0 + 8);
            s0 = wmma_bf16(qa0, kf, s0);
            kf = load_frag16(kr0 + 32, kr0 + 40);
            s0 = wmma_bf16(qa1, kf, s0);
            kf = load_frag16(kr1, kr1 + 8);
            s1 = wmma_bf16(qa0, kf, s1);
            kf = load_frag16(kr1 + 32, kr1 + 40);
            s1 = wmma_bf16(qa1, kf, s1);
        }

        // ---- additive mask + online softmax (reduce over 16-lane halves)
        const float mc0 = maskb[m0 + l16];
        const float mc1 = maskb[m0 + 16 + l16];
        float scale[8];
#pragma unroll
        for (int r = 0; r < 8; ++r) {
            float a = s0[r] - 1.0e8f * (1.0f - mrow[r] * mc0);
            float c = s1[r] - 1.0e8f * (1.0f - mrow[r] * mc1);
            float tmax = fmaxf(a, c);
#pragma unroll
            for (int off = 1; off < 16; off <<= 1)
                tmax = fmaxf(tmax, __shfl_xor(tmax, off, 32));
            const float mn = fmaxf(mmax[r], tmax);
            const float p0 = __expf(a - mn);
            const float p1 = __expf(c - mn);
            float ts = p0 + p1;
#pragma unroll
            for (int off = 1; off < 16; off <<= 1)
                ts += __shfl_xor(ts, off, 32);
            scale[r] = __expf(mmax[r] - mn);
            lsum[r]  = lsum[r] * scale[r] + ts;
            mmax[r]  = mn;
            // P tile [16 rows][32 keys] bf16 in LDS (D-frag -> memory layout)
            pl[(r + half * 8) * 32 + l16]      = f2bf(p0);
            pl[(r + half * 8) * 32 + 16 + l16] = f2bf(p1);
        }

        // ---- rescale running output (row j maps to fragment element j)
#pragma unroll
        for (int r = 0; r < 8; ++r) {
            o0[r] *= scale[r]; o1[r] *= scale[r];
            o2[r] *= scale[r]; o3[r] *= scale[r];
        }

        // wave-local LDS ordering (no cross-wave sharing -> no barrier)
        asm volatile("s_wait_dscnt 0" ::: "memory");

        // ---- P as A-fragment 16x32
        const bf16x16 pa = load_frag16(pl + l16 * 32 + half * 16,
                                       pl + l16 * 32 + 8 + half * 16);

        // ---- O += P @ V   (V transposed: row = s-col, contiguous keys)
        {
            const unsigned short* vr = Vb + (size_t)(0 * 16 + l16) * SEQ + m0 + half * 16;
            bf16x16 vf = load_frag16(vr, vr + 8);
            o0 = wmma_bf16(pa, vf, o0);
            vr = Vb + (size_t)(1 * 16 + l16) * SEQ + m0 + half * 16;
            vf = load_frag16(vr, vr + 8);
            o1 = wmma_bf16(pa, vf, o1);
            vr = Vb + (size_t)(2 * 16 + l16) * SEQ + m0 + half * 16;
            vf = load_frag16(vr, vr + 8);
            o2 = wmma_bf16(pa, vf, o2);
            vr = Vb + (size_t)(3 * 16 + l16) * SEQ + m0 + half * 16;
            vf = load_frag16(vr, vr + 8);
            o3 = wmma_bf16(pa, vf, o3);
        }
    }

    // ---- normalize and store in permuted layout out[b, n, s*H + h]
#pragma unroll
    for (int r = 0; r < 8; ++r) {
        const int n = n0 + r + half * 8;
        const float inv = 1.0f / lsum[r];
        const size_t base = ((size_t)bidx * SEQ + n) * (HSIZE * HEADS) + h;
        out[base + (size_t)(0 * 16 + l16) * HEADS] = o0[r] * inv;
        out[base + (size_t)(1 * 16 + l16) * HEADS] = o1[r] * inv;
        out[base + (size_t)(2 * 16 + l16) * HEADS] = o2[r] * inv;
        out[base + (size_t)(3 * 16 + l16) * HEADS] = o3[r] * inv;
    }
}

// ---------- launch ----------

extern "C" void kernel_launch(void* const* d_in, const int* in_sizes, int n_in,
                              void* d_out, int out_size, void* d_ws, size_t ws_size,
                              hipStream_t stream) {
    const float* x    = (const float*)d_in[0];   // [B,N,D]
    const float* mask = (const float*)d_in[1];   // [B,N]
    const float* w    = (const float*)d_in[2];   // [H,3,D,S]
    float* out        = (float*)d_out;

    char* ws = (char*)d_ws;
    size_t off = 0;
    unsigned short* Xb = (unsigned short*)(ws + off); off += (size_t)BATCH * SEQ * DMODEL * 2;
    unsigned short* Wt = (unsigned short*)(ws + off); off += (size_t)HEADS * 3 * HSIZE * DMODEL * 2;
    unsigned short* Qb = (unsigned short*)(ws + off); off += (size_t)BATCH * HEADS * SEQ * HSIZE * 2;
    unsigned short* Kb = (unsigned short*)(ws + off); off += (size_t)BATCH * HEADS * SEQ * HSIZE * 2;
    unsigned short* Vt = (unsigned short*)(ws + off); off += (size_t)BATCH * HEADS * SEQ * HSIZE * 2;
    (void)ws_size; (void)in_sizes; (void)n_in; (void)out_size;

    cvt_inputs <<<2048, 256, 0, stream>>>(x, Xb, BATCH * SEQ * DMODEL);
    cvt_weights<<<2048, 256, 0, stream>>>(w, Wt);
    qkv_gemm   <<<(256 * 48) / 8, 256, 0, stream>>>(Xb, Wt, Qb, Kb, Vt);
    attn       <<<(BATCH * HEADS * (SEQ / 16)) / 8, 256, 0, stream>>>(Qb, Kb, Vt, mask, out);
}
